// HyperspectralUnmixingModel_87531433492504
// MI455X (gfx1250) — matrix operand order repaired and checked by reading, not compiled, not run on hardware
//
#include <hip/hip_runtime.h>

#define IMG_N   (512*512)
#define NBAND   156
#define NBANDP  160          // K padded to multiple of 32
#define GATHID  64
#define NUME    3
#define MLPHID  128
#define TOTSEG  3500         // 2000 + 1000 + 500

typedef __attribute__((ext_vector_type(16))) _Float16 v16h;
typedef __attribute__((ext_vector_type(8)))  float    v8f;

// ---------------------------------------------------------------- utilities
__global__ void k_zero_f32(float* p, int n){
  int i = blockIdx.x*blockDim.x + threadIdx.x;
  if (i < n) p[i] = 0.0f;
}
__global__ void k_zero_u32(unsigned* p, int n){
  int i = blockIdx.x*blockDim.x + threadIdx.x;
  if (i < n) p[i] = 0u;
}

// float -> order-preserving unsigned encoding (for atomic float max)
__device__ __forceinline__ unsigned fenc(float f){
  unsigned u = __float_as_uint(f);
  return (u & 0x80000000u) ? ~u : (u | 0x80000000u);
}
__device__ __forceinline__ float fdec(unsigned u){
  unsigned b = (u & 0x80000000u) ? (u & 0x7FFFFFFFu) : ~u;
  return __uint_as_float(b);
}

// ------------------------------------------------- segment mean (3 scales)
__global__ void k_seg_accum(const float* __restrict__ Y,
                            const int* __restrict__ s0, const int* __restrict__ s1,
                            const int* __restrict__ s2,
                            float* __restrict__ Xsum, float* __restrict__ cnt){
  int i = blockIdx.x*blockDim.x + threadIdx.x;
  if (i >= IMG_N) return;
  int g0 = s0[i], g1 = 2000 + s1[i], g2 = 3000 + s2[i];
  atomicAdd(&cnt[g0], 1.0f);
  atomicAdd(&cnt[g1], 1.0f);
  atomicAdd(&cnt[g2], 1.0f);
  const float* y = Y + (size_t)i*NBAND;
  float* p0 = Xsum + (size_t)g0*NBAND;
  float* p1 = Xsum + (size_t)g1*NBAND;
  float* p2 = Xsum + (size_t)g2*NBAND;
  for (int c = 0; c < NBAND; ++c){
    float v = y[c];
    atomicAdd(&p0[c], v);
    atomicAdd(&p1[c], v);
    atomicAdd(&p2[c], v);
  }
}
__global__ void k_seg_mean(float* Xsum, const float* cnt, int nodes){
  int i = blockIdx.x*blockDim.x + threadIdx.x;
  if (i >= nodes*NBAND) return;
  Xsum[i] /= fmaxf(cnt[i / NBAND], 1.0f);
}

// --------------------------------------------------- small dense X@W + b
__global__ void k_lin(const float* __restrict__ X, const float* __restrict__ W,
                      const float* __restrict__ b, float* __restrict__ out,
                      int n, int Kin, int Kout){
  int i = blockIdx.x*blockDim.x + threadIdx.x;
  if (i >= n*Kout) return;
  int r = i / Kout, c = i % Kout;
  float acc = b[c];
  const float* x = X + (size_t)r*Kin;
  for (int k = 0; k < Kin; ++k) acc += x[k] * W[(size_t)k*Kout + c];
  out[i] = acc;
}

// --------------------------- fused masked self-attention (one block / row)
template<int D>
__global__ __launch_bounds__(256) void k_gat_attn(const float* __restrict__ Hn,
                                                  const int* __restrict__ A,
                                                  float* __restrict__ out, int n){
  __shared__ float sc[2048];
  __shared__ float hr[GATHID];
  __shared__ float red[256];
  int r = blockIdx.x, tid = threadIdx.x;
  if (tid < D) hr[tid] = Hn[(size_t)r*D + tid];
  __syncthreads();
  float lmax = -1e30f;
  for (int j = tid; j < n; j += 256){
    float s = 0.0f;
    const float* hj = Hn + (size_t)j*D;
#pragma unroll 4
    for (int k = 0; k < D; ++k) s += hr[k]*hj[k];
    s = (A[(size_t)r*n + j] > 0) ? s : -1e9f;
    sc[j] = s;
    lmax = fmaxf(lmax, s);
  }
  red[tid] = lmax; __syncthreads();
  for (int s = 128; s > 0; s >>= 1){
    if (tid < s) red[tid] = fmaxf(red[tid], red[tid + s]);
    __syncthreads();
  }
  float mx = red[0]; __syncthreads();
  float lsum = 0.0f;
  for (int j = tid; j < n; j += 256){
    float e = __expf(sc[j] - mx);
    sc[j] = e; lsum += e;
  }
  red[tid] = lsum; __syncthreads();
  for (int s = 128; s > 0; s >>= 1){
    if (tid < s) red[tid] += red[tid + s];
    __syncthreads();
  }
  float inv = 1.0f / red[0];
  for (int c = tid; c < D; c += 256){
    float acc = 0.0f;
    for (int j = 0; j < n; ++j) acc += sc[j] * Hn[(size_t)j*D + c];
    out[(size_t)r*D + c] = fmaxf(acc * inv, 0.0f);   // relu(aw @ Hn)
  }
}

__global__ void k_relu_softmax3(const float* in, float* out, int n){
  int r = blockIdx.x*blockDim.x + threadIdx.x;
  if (r >= n) return;
  float v0 = fmaxf(in[r*3+0], 0.f), v1 = fmaxf(in[r*3+1], 0.f), v2 = fmaxf(in[r*3+2], 0.f);
  float m = fmaxf(v0, fmaxf(v1, v2));
  float e0 = __expf(v0-m), e1 = __expf(v1-m), e2 = __expf(v2-m);
  float inv = 1.0f/(e0+e1+e2);
  out[r*3+0] = e0*inv; out[r*3+1] = e1*inv; out[r*3+2] = e2*inv;
}

// -------- scatter per-scale abundances to pixels, fuse, start PCR seg-sum
__global__ void k_fuse(const float* __restrict__ sw, const float* __restrict__ ss,
                       const int* __restrict__ s0, const int* __restrict__ s1,
                       const int* __restrict__ s2,
                       float* __restrict__ fused, float* __restrict__ spsum){
  int i = blockIdx.x*blockDim.x + threadIdx.x;
  if (i >= IMG_N) return;
  float w0 = sw[0], w1 = sw[1], w2 = sw[2];
  float m  = fmaxf(w0, fmaxf(w1, w2));
  float e0 = __expf(w0-m), e1 = __expf(w1-m), e2 = __expf(w2-m);
  float inv = 1.0f/(e0+e1+e2); e0 *= inv; e1 *= inv; e2 *= inv;
  int j0 = s0[i];
  const float* p0 = ss + (size_t)j0*3;
  const float* p1 = ss + (size_t)(2000 + s1[i])*3;
  const float* p2 = ss + (size_t)(3000 + s2[i])*3;
#pragma unroll
  for (int e = 0; e < 3; ++e){
    float f = e0*p0[e] + e1*p1[e] + e2*p2[e];
    fused[(size_t)i*3 + e] = f;
    atomicAdd(&spsum[j0*3 + e], f);
  }
}
__global__ void k_spfeat(float* spsum, const float* cnt){
  int i = blockIdx.x*blockDim.x + threadIdx.x;
  if (i >= 2000*3) return;
  spsum[i] /= fmaxf(cnt[i/3], 1.0f);     // cnt[0..1999] is the scale-0 count
}

// --------- row-softmax(attn_mat) @ sp_feat  (one block per superpixel row)
__global__ __launch_bounds__(256) void k_pcr(const float* __restrict__ attn,
                                             const float* __restrict__ spf,
                                             float* __restrict__ sm){
  __shared__ float red[256], r0[256], r1[256], r2[256];
  int r = blockIdx.x, tid = threadIdx.x;
  const float* row = attn + (size_t)r*2000;
  float lmax = -1e30f;
  for (int j = tid; j < 2000; j += 256) lmax = fmaxf(lmax, row[j]);
  red[tid] = lmax; __syncthreads();
  for (int s = 128; s > 0; s >>= 1){
    if (tid < s) red[tid] = fmaxf(red[tid], red[tid+s]);
    __syncthreads();
  }
  float mx = red[0]; __syncthreads();
  float ls = 0.f, a0 = 0.f, a1 = 0.f, a2 = 0.f;
  for (int j = tid; j < 2000; j += 256){
    float e = __expf(row[j] - mx);
    ls += e; a0 += e*spf[j*3]; a1 += e*spf[j*3+1]; a2 += e*spf[j*3+2];
  }
  red[tid]=ls; r0[tid]=a0; r1[tid]=a1; r2[tid]=a2; __syncthreads();
  for (int s = 128; s > 0; s >>= 1){
    if (tid < s){ red[tid]+=red[tid+s]; r0[tid]+=r0[tid+s]; r1[tid]+=r1[tid+s]; r2[tid]+=r2[tid+s]; }
    __syncthreads();
  }
  if (tid == 0){
    float inv = 1.0f/red[0];
    sm[r*3+0]=r0[0]*inv; sm[r*3+1]=r1[0]*inv; sm[r*3+2]=r2[0]*inv;
  }
}

// ------------------- S_flat = a*smoothed + (1-a)*fused ; ci = argmax3
__global__ void k_sflat(const float* __restrict__ fused, const float* __restrict__ sm,
                        const int* __restrict__ s0, const float* __restrict__ alpha,
                        float* __restrict__ S, int* __restrict__ ci){
  int i = blockIdx.x*blockDim.x + threadIdx.x;
  if (i >= IMG_N) return;
  float a = fminf(fmaxf(alpha[0], 0.f), 1.f);
  const float* smr = sm + (size_t)s0[i]*3;
  float v[3];
#pragma unroll
  for (int e = 0; e < 3; ++e){
    v[e] = a*smr[e] + (1.0f-a)*fused[(size_t)i*3+e];
    S[(size_t)i*3+e] = v[e];
  }
  int best = 0;
  if (v[1] > v[0]) best = 1;
  if (v[2] > v[best]) best = 2;
  ci[i] = best;
}

// ----------------------------- WMMA fragment loaders (16x16x32 f16 shapes)
// A 16x32 (M x K): lane holds row M=lane%16, 16 contiguous K values at
//                  kbase + (lane/16)*16.
__device__ __forceinline__ v16h load_a_frag(const _Float16* tile, int ld,
                                            int mbase, int kbase, int lane){
  int m   = mbase + (lane & 15);
  int grp = lane >> 4;
  const _Float16* p = tile + (size_t)m*ld + kbase + grp*16;
  union { v16h v; _Float16 h[16]; } u;
#pragma unroll
  for (int t = 0; t < 16; ++t){ u.h[t] = p[t]; }
  return u.v;
}
// B 32x16 (K x N): lane holds column N=lane%16, 16 contiguous K values at
// kbase + (lane/16)*16.  Weights are pre-transposed to [N][Kpad] f16 so this
// is a single aligned 32-byte load.
__device__ __forceinline__ v16h load_b_frag(const _Float16* WT, int ldk,
                                            int nbase, int kbase, int lane){
  int n   = nbase + (lane & 15);
  int grp = lane >> 4;
  return *(const v16h*)(WT + (size_t)n*ldk + kbase + grp*16);
}

// ---------------- pack MLP weights: f32 [K][N] -> f16 transposed [N][Kpad]
__global__ void k_pack_weights(const float* __restrict__ Wm1, const float* __restrict__ Wm2,
                               const float* __restrict__ Wm3,
                               _Float16* __restrict__ W1T, _Float16* __restrict__ W2T,
                               _Float16* __restrict__ W3T){
  int i = blockIdx.x*blockDim.x + threadIdx.x;
  if (i < MLPHID*NBANDP){                 // W1T [128][160]
    int n = i / NBANDP, k = i % NBANDP;
    W1T[i] = (_Float16)((k < NBAND) ? Wm1[(size_t)k*MLPHID + n] : 0.0f);
  }
  if (i < MLPHID*MLPHID){                 // W2T [128][128]
    int n = i / MLPHID, k = i % MLPHID;
    W2T[i] = (_Float16)Wm2[(size_t)k*MLPHID + n];
  }
  if (i < NBANDP*MLPHID){                 // W3T [160][128], rows >=156 zero
    int n = i / MLPHID, k = i % MLPHID;
    W3T[i] = (_Float16)((n < NBAND) ? Wm3[(size_t)k*NBAND + n] : 0.0f);
  }
}

// --------------- fused 3-layer MLP, 64 pixel-rows per block, WMMA f16->f32
#define MT 64
__global__ __launch_bounds__(256) void k_mlp(const float* __restrict__ Y,
    const _Float16* __restrict__ W1T, const _Float16* __restrict__ W2T,
    const _Float16* __restrict__ W3T,
    const float* __restrict__ b1, const float* __restrict__ b2,
    const float* __restrict__ b3, _Float16* __restrict__ logits){
  __shared__ _Float16 Ah[MT][NBANDP];     // 20 KB: Y tile f16 / layer-2 output
  __shared__ _Float16 Hh[MT][MLPHID];     // 16 KB: layer-1 output
  int tid  = threadIdx.x;
  // wave id is uniform across the 32 lanes; make it scalar so tile loops are
  // SALU-controlled and WMMA always executes with EXEC = all ones.
  int wave = __builtin_amdgcn_readfirstlane(tid >> 5);
  int lane = tid & 31;
  int R = blockIdx.x * MT;

  // stage Y tile -> f16 LDS, and prefetch the next block's tile into L2/L0
  for (int idx = tid; idx < MT*NBANDP; idx += 256){
    int r = idx / NBANDP, c = idx % NBANDP;
    Ah[r][c] = (_Float16)((c < NBAND) ? Y[(size_t)(R + r)*NBAND + c] : 0.0f);
  }
  {
    const float* nxt = Y + (size_t)(R + MT)*NBAND;   // next tile (speculative)
    __builtin_prefetch(nxt + tid*(MT*NBAND/256), 0, 3);
  }
  __syncthreads();

  // layer 1: [64x160] @ [160x128] + b1, relu -> Hh   (4 tiles per wave)
#pragma unroll
  for (int tt = 0; tt < 4; ++tt){
    int t = wave + tt*8;                  // 0..31
    int mt = t >> 3, nt = t & 7;
    v8f acc = {};
#pragma unroll
    for (int kt = 0; kt < NBANDP/32; ++kt){
      v16h a = load_a_frag(&Ah[0][0], NBANDP, mt*16, kt*32, lane);
      v16h b = load_b_frag(W1T, NBANDP, nt*16, kt*32, lane);
      acc = __builtin_amdgcn_wmma_f32_16x16x32_f16(false, a, false, b,
                                                   (short)0, acc, false, false);
    }
    int n  = nt*16 + (lane & 15);
    int mb = mt*16 + (lane >> 4)*8;
    float bias = b1[n];
#pragma unroll
    for (int r2 = 0; r2 < 8; ++r2)
      Hh[mb + r2][n] = (_Float16)fmaxf(acc[r2] + bias, 0.0f);
  }
  __syncthreads();

  // layer 2: [64x128] @ [128x128] + b2, relu -> Ah (ld = NBANDP, cols 0..127)
#pragma unroll
  for (int tt = 0; tt < 4; ++tt){
    int t = wave + tt*8;
    int mt = t >> 3, nt = t & 7;
    v8f acc = {};
#pragma unroll
    for (int kt = 0; kt < MLPHID/32; ++kt){
      v16h a = load_a_frag(&Hh[0][0], MLPHID, mt*16, kt*32, lane);
      v16h b = load_b_frag(W2T, MLPHID, nt*16, kt*32, lane);
      acc = __builtin_amdgcn_wmma_f32_16x16x32_f16(false, a, false, b,
                                                   (short)0, acc, false, false);
    }
    int n  = nt*16 + (lane & 15);
    int mb = mt*16 + (lane >> 4)*8;
    float bias = b2[n];
#pragma unroll
    for (int r2 = 0; r2 < 8; ++r2)
      Ah[mb + r2][n] = (_Float16)fmaxf(acc[r2] + bias, 0.0f);
  }
  __syncthreads();

  // layer 3: [64x128] @ [128x160] + b3 -> logits (store n < 156)
  // 40 tiles, 5 per wave: t = wave + tt*8 covers 0..39 exactly once.
#pragma unroll
  for (int tt = 0; tt < 5; ++tt){
    int t = wave + tt*8;
    int mt = t / 10, nt = t % 10;
    v8f acc = {};
#pragma unroll
    for (int kt = 0; kt < MLPHID/32; ++kt){
      v16h a = load_a_frag(&Ah[0][0], NBANDP, mt*16, kt*32, lane);
      v16h b = load_b_frag(W3T, MLPHID, nt*16, kt*32, lane);
      acc = __builtin_amdgcn_wmma_f32_16x16x32_f16(false, a, false, b,
                                                   (short)0, acc, false, false);
    }
    int n  = nt*16 + (lane & 15);
    int mb = mt*16 + (lane >> 4)*8;
    if (n < NBAND){
      float bias = b3[n];
#pragma unroll
      for (int r2 = 0; r2 < 8; ++r2)
        logits[(size_t)(R + mb + r2)*NBAND + n] = (_Float16)(acc[r2] + bias);
    }
  }
}

// ------------------- per-(class,band) max of logits over masked rows
__global__ __launch_bounds__(256) void k_class_max(const _Float16* __restrict__ logits,
                                                   const int* __restrict__ ci,
                                                   unsigned* __restrict__ encMax){
  __shared__ unsigned lmax[NUME*NBAND];
  int tid = threadIdx.x;
  for (int i = tid; i < NUME*NBAND; i += 256) lmax[i] = 0u;
  __syncthreads();
  int row = blockIdx.x*256 + tid;
  if (row < IMG_N){
    int p = ci[row];
    const _Float16* lr = logits + (size_t)row*NBAND;
    for (int c = 0; c < NBAND; ++c)
      atomicMax(&lmax[p*NBAND + c], fenc((float)lr[c]));
  }
  __syncthreads();
  for (int i = tid; i < NUME*NBAND; i += 256)
    if (lmax[i]) atomicMax(&encMax[i], lmax[i]);
}

// ------------- per-(class,band) sum(exp) and sum(exp*Y) over masked rows
__global__ __launch_bounds__(256) void k_class_sum(const _Float16* __restrict__ logits,
                                                   const int* __restrict__ ci,
                                                   const float* __restrict__ Y,
                                                   const unsigned* __restrict__ encMax,
                                                   float* __restrict__ sumE,
                                                   float* __restrict__ sumEY){
  __shared__ float lsE[NUME*NBAND], lsEY[NUME*NBAND], mx[NUME*NBAND];
  int tid = threadIdx.x;
  for (int i = tid; i < NUME*NBAND; i += 256){
    lsE[i] = 0.f; lsEY[i] = 0.f;
    unsigned e = encMax[i];
    mx[i] = e ? fdec(e) : 0.0f;
  }
  __syncthreads();
  int row = blockIdx.x*256 + tid;
  if (row < IMG_N){
    int p = ci[row];
    const _Float16* lr = logits + (size_t)row*NBAND;
    const float*    yr = Y + (size_t)row*NBAND;
    for (int c = 0; c < NBAND; ++c){
      float e = __expf((float)lr[c] - mx[p*NBAND + c]);
      atomicAdd(&lsE [p*NBAND + c], e);
      atomicAdd(&lsEY[p*NBAND + c], e*yr[c]);
    }
  }
  __syncthreads();
  for (int i = tid; i < NUME*NBAND; i += 256){
    if (lsE[i] != 0.0f){
      atomicAdd(&sumE[i],  lsE[i]);
      atomicAdd(&sumEY[i], lsEY[i]);
    }
  }
}

__global__ void k_endmembers(const float* sumE, const float* sumEY, float* M){
  int i = blockIdx.x*blockDim.x + threadIdx.x;
  if (i >= NUME*NBAND) return;
  float d = sumE[i];
  M[i] = (d > 0.0f) ? sumEY[i] / fmaxf(d, 1e-30f) : 0.0f;
}

// ------------------------------- Y_hat = S_flat [N,3] @ M [3,156]
__global__ void k_reconstruct(const float* __restrict__ S, const float* __restrict__ M,
                              float* __restrict__ out){
  long long i = (long long)blockIdx.x*blockDim.x + threadIdx.x;
  if (i >= (long long)IMG_N*NBAND) return;
  int r = (int)(i / NBAND), c = (int)(i % NBAND);
  out[i] = S[(size_t)r*3+0]*M[c] + S[(size_t)r*3+1]*M[NBAND+c]
         + S[(size_t)r*3+2]*M[2*NBAND+c];
}

// ======================================================================
extern "C" void kernel_launch(void* const* d_in, const int* in_sizes, int n_in,
                              void* d_out, int out_size, void* d_ws, size_t ws_size,
                              hipStream_t stream) {
  (void)in_sizes; (void)n_in; (void)out_size; (void)ws_size;
  // dict order from setup_inputs():
  const float* Y    = (const float*)d_in[0];
  const int*   s0   = (const int*)  d_in[1];
  const int*   A0   = (const int*)  d_in[2];
  const int*   s1   = (const int*)  d_in[3];
  const int*   A1   = (const int*)  d_in[4];
  const int*   s2   = (const int*)  d_in[5];
  const int*   A2   = (const int*)  d_in[6];
  const float* attn = (const float*)d_in[7];
  const float* alpha= (const float*)d_in[8];
  const float* sw   = (const float*)d_in[9];
  const float* W1[3] = {(const float*)d_in[10], (const float*)d_in[14], (const float*)d_in[18]};
  const float* b1[3] = {(const float*)d_in[11], (const float*)d_in[15], (const float*)d_in[19]};
  const float* W2[3] = {(const float*)d_in[12], (const float*)d_in[16], (const float*)d_in[20]};
  const float* b2[3] = {(const float*)d_in[13], (const float*)d_in[17], (const float*)d_in[21]};
  const float* Wm1 = (const float*)d_in[22];
  const float* bm1 = (const float*)d_in[23];
  const float* Wm2 = (const float*)d_in[24];
  const float* bm2 = (const float*)d_in[25];
  const float* Wm3 = (const float*)d_in[26];
  const float* bm3 = (const float*)d_in[27];
  float* out = (float*)d_out;

  // ---- workspace bump allocator (~92 MB total) ----
  char* w = (char*)d_ws; size_t off = 0;
  auto alloc = [&](size_t bytes) -> void* {
    off = (off + 255) & ~(size_t)255;
    void* p = w + off; off += bytes; return p;
  };
  float*     Xsum  = (float*)alloc((size_t)TOTSEG*NBAND*4);
  float*     cnt   = (float*)alloc((size_t)TOTSEG*4);
  float*     HnA   = (float*)alloc((size_t)2000*GATHID*4);
  float*     HnB   = (float*)alloc((size_t)2000*GATHID*4);
  float*     Hs3a  = (float*)alloc((size_t)2000*3*4);
  float*     Hs3b  = (float*)alloc((size_t)2000*3*4);
  float*     sscale= (float*)alloc((size_t)TOTSEG*3*4);
  float*     fused = (float*)alloc((size_t)IMG_N*3*4);
  float*     spsum = (float*)alloc((size_t)2000*3*4);
  float*     smsp  = (float*)alloc((size_t)2000*3*4);
  float*     Sflat = (float*)alloc((size_t)IMG_N*3*4);
  int*       ci    = (int*)  alloc((size_t)IMG_N*4);
  _Float16*  logit = (_Float16*)alloc((size_t)IMG_N*NBAND*2);
  _Float16*  W1T   = (_Float16*)alloc((size_t)MLPHID*NBANDP*2);
  _Float16*  W2T   = (_Float16*)alloc((size_t)MLPHID*MLPHID*2);
  _Float16*  W3T   = (_Float16*)alloc((size_t)NBANDP*MLPHID*2);
  unsigned*  encMx = (unsigned*)alloc((size_t)NUME*NBAND*4);
  float*     sumE  = (float*)alloc((size_t)NUME*NBAND*4);
  float*     sumEY = (float*)alloc((size_t)NUME*NBAND*4);
  float*     Mmat  = (float*)alloc((size_t)NUME*NBAND*4);

  auto cdiv = [](int a, int b){ return (a + b - 1) / b; };

  // ---- zero accumulators (fresh every call) ----
  k_zero_f32<<<cdiv(TOTSEG*NBAND,256),256,0,stream>>>(Xsum, TOTSEG*NBAND);
  k_zero_f32<<<cdiv(TOTSEG,256),256,0,stream>>>(cnt, TOTSEG);
  k_zero_f32<<<cdiv(2000*3,256),256,0,stream>>>(spsum, 2000*3);
  k_zero_f32<<<cdiv(NUME*NBAND,256),256,0,stream>>>(sumE, NUME*NBAND);
  k_zero_f32<<<cdiv(NUME*NBAND,256),256,0,stream>>>(sumEY, NUME*NBAND);
  k_zero_u32<<<cdiv(NUME*NBAND,256),256,0,stream>>>(encMx, NUME*NBAND);

  // ---- pack MLP weights to f16 transposed ----
  k_pack_weights<<<cdiv(NBANDP*MLPHID,256),256,0,stream>>>(Wm1, Wm2, Wm3, W1T, W2T, W3T);

  // ---- segment means (all 3 scales) ----
  k_seg_accum<<<cdiv(IMG_N,256),256,0,stream>>>(Y, s0, s1, s2, Xsum, cnt);
  k_seg_mean<<<cdiv(TOTSEG*NBAND,256),256,0,stream>>>(Xsum, cnt, TOTSEG);

  // ---- multi-scale GAT encoder ----
  const int   segn[3]  = {2000, 1000, 500};
  const int   segoff[3]= {0, 2000, 3000};
  const int*  Amat[3]  = {A0, A1, A2};
  for (int s = 0; s < 3; ++s){
    int n = segn[s];
    const float* Xs = Xsum + (size_t)segoff[s]*NBAND;
    k_lin<<<cdiv(n*GATHID,256),256,0,stream>>>(Xs, W1[s], b1[s], HnA, n, NBAND, GATHID);
    k_gat_attn<GATHID><<<n,256,0,stream>>>(HnA, Amat[s], HnB, n);
    k_lin<<<cdiv(n*3,256),256,0,stream>>>(HnB, W2[s], b2[s], Hs3a, n, GATHID, NUME);
    k_gat_attn<NUME><<<n,256,0,stream>>>(Hs3a, Amat[s], Hs3b, n);
    k_relu_softmax3<<<cdiv(n,256),256,0,stream>>>(Hs3b, sscale + (size_t)segoff[s]*3, n);
  }

  // ---- fuse scales to pixels + PCR superpixel sums ----
  k_fuse<<<cdiv(IMG_N,256),256,0,stream>>>(sw, sscale, s0, s1, s2, fused, spsum);
  k_spfeat<<<cdiv(2000*3,256),256,0,stream>>>(spsum, cnt);
  k_pcr<<<2000,256,0,stream>>>(attn, spsum, smsp);
  k_sflat<<<cdiv(IMG_N,256),256,0,stream>>>(fused, smsp, s0, alpha, Sflat, ci);

  // ---- ACDE: fused WMMA MLP -> logits, then class-masked softmax stats ----
  k_mlp<<<IMG_N/MT,256,0,stream>>>(Y, W1T, W2T, W3T, bm1, bm2, bm3, logit);
  k_class_max<<<cdiv(IMG_N,256),256,0,stream>>>(logit, ci, encMx);
  k_class_sum<<<cdiv(IMG_N,256),256,0,stream>>>(logit, ci, Y, encMx, sumE, sumEY);
  k_endmembers<<<cdiv(NUME*NBAND,256),256,0,stream>>>(sumE, sumEY, Mmat);

  // ---- Y_hat = S_flat @ M ----
  long long tot = (long long)IMG_N*NBAND;
  k_reconstruct<<<(int)((tot + 255)/256),256,0,stream>>>(Sflat, Mmat, out);
}
